// ShiftWindowMSA_26199300505805
// MI455X (gfx1250) — matrix operand order repaired and checked by reading, not compiled, not run on hardware
//
#include <hip/hip_runtime.h>

typedef __attribute__((ext_vector_type(16))) _Float16 v16h;
typedef __attribute__((ext_vector_type(8)))  _Float16 v8h;
typedef __attribute__((ext_vector_type(8)))  float    v8f;

#define IMG   224
#define CH    96
#define NHEAD 3
#define HD    32
#define WSZ   7
#define NTOK  49
#define SHFT  3
#define LTOK  (IMG*IMG)      /* 50176 */
#define BATCH 4
#define MTOT  (BATCH*LTOK)   /* 200704 */

__device__ __forceinline__ int band7(int p) {
  // region bands for the shifted-window attention mask (Hp=Wp=224, w=7, s=3)
  return p < (IMG - WSZ) ? 0 : (p < (IMG - SHFT) ? 1 : 2);
}

// A-fragment for v_wmma_f32_16x16x32_f16 from row-major f16 (row=M, K contiguous).
// a[e] <- base[row*stride + col + (lane/16)*8 + e%8 + 16*(e/8)]
__device__ __forceinline__ v16h load_a_frag(const _Float16* base, int stride,
                                            int row, int col, int lh) {
  v8h x1 = *(const v8h*)(base + row * stride + col + lh * 8);
  v8h x2 = *(const v8h*)(base + row * stride + col + 16 + lh * 8);
  v16h a;
#pragma unroll
  for (int e = 0; e < 8; ++e) { a[e] = x1[e]; a[e + 8] = x2[e]; }
  return a;
}

// B-fragment (K x 16) from a row-major f16 matrix whose rows are indexed by n with
// K contiguous (i.e. B^T layout): b[e] <- base[nrow*stride + col + (lane/16)*8 + e%8 + 16*(e/8)]
__device__ __forceinline__ v16h load_b_frag_kcontig(const _Float16* base, int stride,
                                                    int nrow, int col, int lh) {
  v8h y1 = *(const v8h*)(base + nrow * stride + col + lh * 8);
  v8h y2 = *(const v8h*)(base + nrow * stride + col + 16 + lh * 8);
  v16h b;
#pragma unroll
  for (int e = 0; e < 8; ++e) { b[e] = y1[e]; b[e + 8] = y2[e]; }
  return b;
}

// ---------------------------------------------------------------------------
// Kernel 0: one-time f32 -> f16 conversion of the small weight matrices.
// ---------------------------------------------------------------------------
__global__ __launch_bounds__(256) void cvt_f32_f16(const float* __restrict__ src,
                                                   _Float16* __restrict__ dst, int n) {
  int i = blockIdx.x * 256 + threadIdx.x;
  if (i < n) dst[i] = (_Float16)src[i];
}

// ---------------------------------------------------------------------------
// Kernel 1: Y[M,N] (f16) = X[M,96] (f32) @ W16[N,96]^T + bias   (token linear)
// One block = 16 rows of X staged in LDS as f16; 4 waves sweep all N tiles.
// ---------------------------------------------------------------------------
__global__ __launch_bounds__(128) void linproj_f16(
    const float* __restrict__ X, const _Float16* __restrict__ W16,
    const float* __restrict__ bias, _Float16* __restrict__ Y, int N) {
  __shared__ __align__(16) _Float16 sA[16 * CH];
  const int tid = threadIdx.x;
  const int m0  = blockIdx.x * 16;

  // pull weight cachelines toward the WGP early (global_prefetch_b8)
  __builtin_prefetch(W16 + (size_t)tid * ((size_t)N * CH / 128), 0, 1);

#pragma unroll 4
  for (int idx = tid; idx < 16 * CH; idx += 128) {
    sA[idx] = (_Float16)X[(size_t)m0 * CH + idx];
  }
  __syncthreads();

  const int wave = tid >> 5, lane = tid & 31;
  const int lh = lane >> 4, ln = lane & 15;

  for (int n0 = wave * 16; n0 < N; n0 += 64) {
    const int n = n0 + ln;
    v8f acc = {};
#pragma unroll
    for (int kb = 0; kb < CH; kb += 32) {
      v16h a = load_a_frag(sA, CH, ln, kb, lh);
      v16h b = load_b_frag_kcontig(W16, CH, n, kb, lh);
      acc = __builtin_amdgcn_wmma_f32_16x16x32_f16(false, a, false, b,
                                                   (short)0, acc, false, false);
    }
    const float bv = bias[n];
#pragma unroll
    for (int r = 0; r < 8; ++r) {
      Y[(size_t)(m0 + r + lh * 8) * N + n] = (_Float16)(acc[r] + bv);
    }
  }
}

// ---------------------------------------------------------------------------
// Kernel 2: per-window attention + proj + scatter. 1 block (8 wave32) / window.
// ---------------------------------------------------------------------------
__device__ __forceinline__ size_t src_token(int bImg, int wr, int wc, int t) {
  int i = t / WSZ, j = t - i * WSZ;
  int sh = wr * WSZ + i + SHFT; if (sh >= IMG) sh -= IMG;  // roll(-s) source
  int sw = wc * WSZ + j + SHFT; if (sw >= IMG) sw -= IMG;
  return (size_t)bImg * LTOK + (size_t)sh * IMG + sw;
}

__global__ __launch_bounds__(256) void swin_attn(
    const _Float16* __restrict__ kvbuf,   // (M,192): [0,96)=k, [96,192)=v
    const _Float16* __restrict__ qbuf,    // (M,96), unscaled (1/sqrt(hd) applied at gather)
    const float* __restrict__ relTab,     // (169,3)
    const _Float16* __restrict__ projW16, // (96,96) f16
    const float* __restrict__ projB,      // (96)
    float* __restrict__ out) {
  __shared__ __align__(16) char smem[4 * 64 * CH * 2 + 64 * 64 * 4];  // 64 KB
  _Float16* sQ  = (_Float16*)smem;       // 64 x 96
  _Float16* sK  = sQ  + 64 * CH;         // 64 x 96
  _Float16* sVT = sK  + 64 * CH;         // 96 x 64  (V transposed: [dim][token])
  _Float16* sO  = sVT + CH * 64;         // 64 x 96
  float*    sS  = (float*)(smem + 4 * 64 * CH * 2);  // 64 x 64 f32 scores
  _Float16* sP  = (_Float16*)sS;         // f16 probs overlay, row stride 128

  const int tid  = threadIdx.x;
  const int lane = tid & 31, wave = tid >> 5;
  const int lh = lane >> 4, ln = lane & 15;
  const int wid  = blockIdx.x;
  const int bImg = wid >> 10;
  const int win  = wid & 1023;
  const int wr = win >> 5, wc = win & 31;

  // warm the projection weights (global_prefetch_b8); 96*96*2 B / 256 thr = 72 B apart
  __builtin_prefetch(projW16 + (size_t)tid * 36, 0, 1);

  // ---- gather K (row-major) and V (transposed) ----
  for (int idx = tid; idx < NTOK * 24; idx += 256) {
    int t = idx / 24, chv = idx - t * 24;
    size_t row = src_token(bImg, wr, wc, t);
    v8h val = *(const v8h*)(kvbuf + row * (2 * CH) + chv * 8);
    if (chv < 12) {
      *(v8h*)(sK + t * CH + chv * 8) = val;
    } else {
      int d0 = (chv - 12) * 8;
#pragma unroll
      for (int e = 0; e < 8; ++e) sVT[(d0 + e) * 64 + t] = val[e];
    }
  }
  // ---- gather Q with 1/sqrt(hd) folded in ----
  for (int idx = tid; idx < NTOK * 12; idx += 256) {
    int t = idx / 12, chv = idx - t * 12;
    size_t row = src_token(bImg, wr, wc, t);
    v8h val = *(const v8h*)(qbuf + row * CH + chv * 8);
    v8h sv;
#pragma unroll
    for (int e = 0; e < 8; ++e)
      sv[e] = (_Float16)((float)val[e] * 0.17677669529663687f);
    *(v8h*)(sQ + t * CH + chv * 8) = sv;
  }
  // ---- zero pads: Q/K rows 49..63, V^T token columns 49..63 ----
  for (int idx = tid; idx < (64 - NTOK) * CH; idx += 256) {
    int off = NTOK * CH + idx;
    sQ[off] = (_Float16)0.f;
    sK[off] = (_Float16)0.f;
    int d = idx / (64 - NTOK), t = NTOK + (idx - d * (64 - NTOK));
    sVT[d * 64 + t] = (_Float16)0.f;
  }

  for (int h = 0; h < NHEAD; ++h) {
    __syncthreads();
    // ---- S = Q_h @ K_h^T  (4x4 tiles of 16x16, K=32 -> one wmma each) ----
    for (int tile = wave; tile < 16; tile += 8) {
      int mt = tile >> 2, nt = tile & 3;
      v16h a = load_a_frag(sQ, CH, mt * 16 + ln, h * HD, lh);
      v16h b = load_b_frag_kcontig(sK, CH, nt * 16 + ln, h * HD, lh);
      v8f acc = {};
      acc = __builtin_amdgcn_wmma_f32_16x16x32_f16(false, a, false, b,
                                                   (short)0, acc, false, false);
#pragma unroll
      for (int r = 0; r < 8; ++r)
        sS[(mt * 16 + r + lh * 8) * 64 + nt * 16 + ln] = acc[r];
    }
    __syncthreads();
    // ---- softmax with rel-pos bias + shift mask; in-place f32 -> f16 ----
    if (tid < NTOK) {
      const int ar   = tid;
      const int va   = band7(wr * WSZ + ar / WSZ) * 3 + band7(wc * WSZ + ar % WSZ);
      const int relA = 13 * (ar / WSZ) + ar % WSZ;
      float* srow = sS + ar * 64;
      float mx = -1e30f;
      for (int c = 0; c < NTOK; ++c) {
        int fb   = 48 - c;                                   // flip along dim 1
        int ridx = relA + 13 * (fb / WSZ) + fb % WSZ;        // < 169
        float bv = relTab[ridx * NHEAD + h];
        int vb   = band7(wr * WSZ + c / WSZ) * 3 + band7(wc * WSZ + c % WSZ);
        float s  = srow[c] + bv + ((va != vb) ? -100.0f : 0.0f);
        srow[c] = s;
        mx = fmaxf(mx, s);
      }
      float sum = 0.f;
      for (int c = 0; c < NTOK; ++c) {
        float e = __expf(srow[c] - mx);
        srow[c] = e;
        sum += e;
      }
      float inv = 1.0f / sum;
      // forward-order compaction: write f16 at byte 2c after reading f32 at 4c
      volatile float*    vsrc = srow;
      volatile _Float16* vdst = (volatile _Float16*)srow;
      for (int c = 0; c < NTOK; ++c) {
        float e = vsrc[c];
        vdst[c] = (_Float16)(e * inv);
      }
      for (int c = NTOK; c < 64; ++c) vdst[c] = (_Float16)0.f;  // zero K-pad cols
    }
    __syncthreads();
    // ---- O_h = P @ V_h  (4x2 tiles, K=64 -> two wmma) : 8 tasks = 8 waves ----
    {
      int mt = wave >> 1, nt = wave & 1;
      v8f acc = {};
#pragma unroll
      for (int kb = 0; kb < 64; kb += 32) {
        v16h a = load_a_frag(sP, 128, mt * 16 + ln, kb, lh);
        // V^T is [dim][token] with token contiguous -> K-contiguous B fragment
        v16h b = load_b_frag_kcontig(sVT, 64, h * HD + nt * 16 + ln - ln, 0, 0);
        // (the helper expects nrow indexing; build explicitly instead)
        b = load_b_frag_kcontig(sVT, 64, h * HD + nt * 16 + ln, kb, lh);
        acc = __builtin_amdgcn_wmma_f32_16x16x32_f16(false, a, false, b,
                                                     (short)0, acc, false, false);
      }
#pragma unroll
      for (int r = 0; r < 8; ++r)
        sO[(mt * 16 + r + lh * 8) * CH + h * HD + nt * 16 + ln] = (_Float16)acc[r];
    }
  }
  __syncthreads();

  // ---- final projection + reverse-shift scatter ----
  for (int task = wave; task < 24; task += 8) {
    int mt = task / 6, nt = task - (task / 6) * 6;
    int n  = nt * 16 + ln;
    v8f acc = {};
#pragma unroll
    for (int kb = 0; kb < CH; kb += 32) {
      v16h a = load_a_frag(sO, CH, mt * 16 + ln, kb, lh);
      v16h b = load_b_frag_kcontig(projW16, CH, n, kb, lh);
      acc = __builtin_amdgcn_wmma_f32_16x16x32_f16(false, a, false, b,
                                                   (short)0, acc, false, false);
    }
    float pb = projB[n];
#pragma unroll
    for (int r = 0; r < 8; ++r) {
      int m = mt * 16 + r + lh * 8;
      if (m < NTOK) {
        int i = m / WSZ, j = m - i * WSZ;
        int dh = wr * WSZ + i + SHFT; if (dh >= IMG) dh -= IMG;  // roll(+s) dest
        int dw = wc * WSZ + j + SHFT; if (dw >= IMG) dw -= IMG;
        out[((size_t)bImg * LTOK + (size_t)dh * IMG + dw) * CH + n] = acc[r] + pb;
      }
    }
  }
}

extern "C" void kernel_launch(void* const* d_in, const int* in_sizes, int n_in,
                              void* d_out, int out_size, void* d_ws, size_t ws_size,
                              hipStream_t stream) {
  (void)in_sizes; (void)n_in; (void)out_size; (void)ws_size;
  const float* query  = (const float*)d_in[0];
  const float* skipq  = (const float*)d_in[1];
  const float* qkv_w  = (const float*)d_in[2];
  const float* qkv_b  = (const float*)d_in[3];
  const float* skip_w = (const float*)d_in[4];
  const float* skip_b = (const float*)d_in[5];
  const float* proj_w = (const float*)d_in[6];
  const float* proj_b = (const float*)d_in[7];
  const float* relTab = (const float*)d_in[8];
  float* out = (float*)d_out;

  // workspace carve-up (all 16B aligned)
  _Float16* kvbuf = (_Float16*)d_ws;                                    // 200704 x 192
  _Float16* qbuf  = kvbuf + (size_t)MTOT * (2 * CH);                    // 200704 x 96
  _Float16* qkvW16  = qbuf + (size_t)MTOT * CH;                         // 192 x 96
  _Float16* skipW16 = qkvW16  + (size_t)(2 * CH) * CH;                  // 96 x 96
  _Float16* projW16 = skipW16 + (size_t)CH * CH;                        // 96 x 96

  cvt_f32_f16<<<dim3((2 * CH * CH + 255) / 256), dim3(256), 0, stream>>>(
      qkv_w, qkvW16, 2 * CH * CH);
  cvt_f32_f16<<<dim3((CH * CH + 255) / 256), dim3(256), 0, stream>>>(
      skip_w, skipW16, CH * CH);
  cvt_f32_f16<<<dim3((CH * CH + 255) / 256), dim3(256), 0, stream>>>(
      proj_w, projW16, CH * CH);

  dim3 blk(128);
  dim3 grd(MTOT / 16);
  linproj_f16<<<grd, blk, 0, stream>>>(query, qkvW16, qkv_b, kvbuf, 2 * CH);
  linproj_f16<<<grd, blk, 0, stream>>>(skipq, skipW16, skip_b, qbuf, CH);

  swin_attn<<<dim3(BATCH * 1024), dim3(256), 0, stream>>>(
      kvbuf, qbuf, relTab, projW16, proj_b, out);
}
